// Block_74431783239935
// MI455X (gfx1250) — matrix operand order repaired and checked by reading, not compiled, not run on hardware
//
#include <hip/hip_runtime.h>
#include <math.h>
#include <stdint.h>

typedef __attribute__((ext_vector_type(16))) __bf16 v16bf;
typedef __attribute__((ext_vector_type(8)))  float  v8f;
typedef __attribute__((ext_vector_type(8)))  int    v8i;
typedef __attribute__((ext_vector_type(4)))  int    v4i;

// ---- async global->LDS copy (CDNA5) with synchronous fallback ----
#if defined(__has_builtin)
#  if __has_builtin(__builtin_amdgcn_global_load_async_to_lds_b128)
#    define USE_ASYNC_CP 1
#  endif
#endif
#ifndef USE_ASYNC_CP
#  define USE_ASYNC_CP 0
#endif

typedef __attribute__((address_space(1))) v4i gas_v4i;   // global (AS1) int4
typedef __attribute__((address_space(3))) v4i las_v4i;   // LDS (AS3) int4

__device__ __forceinline__ void cp16(void* lds, const void* gptr) {
#if USE_ASYNC_CP
  __builtin_amdgcn_global_load_async_to_lds_b128(
      (gas_v4i*)(uintptr_t)gptr,
      (las_v4i*)(unsigned)(uintptr_t)lds,
      0, 0);
#else
  *(v4i*)lds = *(const v4i*)gptr;
#endif
}

__device__ __forceinline__ void wait_async_all() {
#if USE_ASYNC_CP
#  if __has_builtin(__builtin_amdgcn_s_wait_asynccnt)
  __builtin_amdgcn_s_wait_asynccnt(0);
#  else
  asm volatile("s_wait_asynccnt 0" ::: "memory");
#  endif
#endif
}

// ---- WMMA fragment index helpers (wave32, 16x16x32 bf16) ----
// A-matrix 16x32 (ISA 7.12.2): lanes 0-15 hold K={0..7,16..23}, lanes 16-31 K={8..15,24..31}
__device__ __forceinline__ int fdA(int v, int hi) {
  return (v & 3) + ((v >> 2) << 3) + (hi ? 4 : 0);   // dword index in 16-dword K-chunk
}
// B-matrix 32x16: same K layout as A: lanes 0-15 hold K={0..7,16..23}, lanes 16-31 K={8..15,24..31}
__device__ __forceinline__ int fdB(int v, int hi) {
  return (v & 3) + ((v >> 2) << 3) + (hi ? 4 : 0);
}

__device__ __forceinline__ v8f wmma_bf16(v8i a, v8i b, v8f c) {
  return __builtin_amdgcn_wmma_f32_16x16x32_bf16(
      false, __builtin_bit_cast(v16bf, a),
      false, __builtin_bit_cast(v16bf, b),
      (short)0, c, false, false);
}

// ---------------- LayerNorm: fp32 [rows][C] -> bf16 [rows][Cp] (zero pad) ----------------
__global__ __launch_bounds__(256) void layernorm_kernel(
    const float* __restrict__ x, const float* __restrict__ g, const float* __restrict__ bta,
    __bf16* __restrict__ out, int C, int Cp)
{
  __shared__ float s1[256];
  __shared__ float s2[256];
  const int tid = threadIdx.x;
  const long row = blockIdx.x;
  const float* xr = x + row * C;
  float s = 0.f, q = 0.f;
  for (int c = tid; c < C; c += 256) { float v = xr[c]; s += v; q += v * v; }
  s1[tid] = s; s2[tid] = q;
  __syncthreads();
  for (int off = 128; off > 0; off >>= 1) {
    if (tid < off) { s1[tid] += s1[tid + off]; s2[tid] += s2[tid + off]; }
    __syncthreads();
  }
  const float mu  = s1[0] / (float)C;
  const float var = s2[0] / (float)C - mu * mu;
  const float inv = rsqrtf(var + 1e-5f);
  __bf16* orow = out + row * Cp;
  for (int c = tid; c < Cp; c += 256) {
    float v = 0.f;
    if (c < C) v = (xr[c] - mu) * inv * g[c] + bta[c];
    orow[c] = (__bf16)v;
  }
}

// ------------- Pad + transpose convert: fp32 src [R][C] -> bf16 dst [Cp][Rp] -------------
// dst[c][r] = src[r][c]  (weights become [N][K] so GEMM staging is K-contiguous byte copy)
__global__ __launch_bounds__(256) void padconvT_kernel(
    const float* __restrict__ src, __bf16* __restrict__ dst,
    int R, int C, int Rp, int Cp)
{
  long i = (long)blockIdx.x * 256 + threadIdx.x;
  long total = (long)Rp * Cp;
  if (i >= total) return;
  int c = (int)(i / Rp), r = (int)(i % Rp);
  float v = (r < R && c < C) ? src[(long)r * C + c] : 0.f;
  dst[i] = (__bf16)v;
}

// ---- Pack Wq/Wk/Wv [6][684][114] fp32 -> transposed [2304][704] bf16 (row n = out col) ----
// n in [0,768)=Q heads (128 each), [768,1536)=K, [1536,2304)=V; pads zero.
__global__ __launch_bounds__(256) void packqkvT_kernel(
    const float* __restrict__ Wq, const float* __restrict__ Wk, const float* __restrict__ Wv,
    __bf16* __restrict__ dst)
{
  long i = (long)blockIdx.x * 256 + threadIdx.x;   // exactly 2304*704 threads
  int c = (int)(i / 704), d = (int)(i % 704);
  int s = c / 768, w = c % 768, hh = w >> 7, e = w & 127;
  float v = 0.f;
  if (d < 684 && e < 114) {
    const float* W = (s == 0) ? Wq : ((s == 1) ? Wk : Wv);
    v = W[((long)hh * 684 + d) * 114 + e];
  }
  dst[i] = (__bf16)v;
}

// ------- bf16 WMMA GEMM: C = A[MxK] * Bt^T (Bt is [N][K]) (+bias,relu,resid) -------
// Block tile 256x64, 256 threads (8 waves), each wave 32 rows x 64 cols.
// Double-buffered LDS, async global->LDS staging, one barrier per K-step.
template<bool HASBIAS, bool RELU, bool RESID, bool OUTF32>
__global__ __launch_bounds__(256) void gemm_bf16_kernel(
    const __bf16* __restrict__ A, const __bf16* __restrict__ Bt,
    const float* __restrict__ bias, const float* __restrict__ resid,
    float* __restrict__ outF, __bf16* __restrict__ outB,
    int M, int N, int K, int nvalid)
{
  __shared__ __attribute__((aligned(16))) __bf16 sA[2][256 * 32];  // 16KB each
  __shared__ __attribute__((aligned(16))) __bf16 sB[2][64 * 32];   // 4KB each
  const int tid  = threadIdx.x;
  const int lane = tid & 31, wave = tid >> 5, hi = lane >> 4;
  const long row0 = (long)blockIdx.x * 256;
  const int  col0 = blockIdx.y * 64;
  const char* Ab = (const char*)A;
  const char* Bb = (const char*)Bt;
  const long ldab = (long)K * 2;   // bytes per A row
  const long ldbb = (long)K * 2;   // bytes per Bt row

  auto stage = [&](int kstep, int buf) {
    const long koff = (long)kstep * 64;              // 32 bf16 = 64 bytes
    char* a_l = (char*)sA[buf];
    char* b_l = (char*)sB[buf];
#pragma unroll
    for (int i = 0; i < 4; ++i) {                    // A: 256 rows x 4 chunks of 16B
      int idx = tid + (i << 8);
      int r = idx >> 2, c = idx & 3;
      cp16(a_l + idx * 16, Ab + (row0 + r) * ldab + koff + c * 16);
    }
    {                                                // B: 64 rows x 4 chunks of 16B
      int r = tid >> 2, c = tid & 3;
      cp16(b_l + tid * 16, Bb + (long)(col0 + r) * ldbb + koff + c * 16);
    }
  };

  v8f acc[2][4] = {};
  const int nk = K >> 5;
  stage(0, 0);
  for (int i = 0; i < nk; ++i) {
    wait_async_all();
    __syncthreads();              // all waves: group i landed, buf[(i+1)&1] free
    if (i + 1 < nk) stage(i + 1, (i + 1) & 1);
    const int* sAi = (const int*)sA[i & 1];
    const int* sBi = (const int*)sB[i & 1];
    v8i a0, a1;
    const int ab0 = (wave * 32 + (lane & 15)) * 16;
    const int ab1 = ab0 + 256;
#pragma unroll
    for (int v = 0; v < 8; ++v) { a0[v] = sAi[ab0 + fdA(v, hi)]; a1[v] = sAi[ab1 + fdA(v, hi)]; }
#pragma unroll
    for (int nt = 0; nt < 4; ++nt) {
      v8i b;
      const int bb = (nt * 16 + (lane & 15)) * 16;
#pragma unroll
      for (int v = 0; v < 8; ++v) b[v] = sBi[bb + fdB(v, hi)];
      acc[0][nt] = wmma_bf16(a0, b, acc[0][nt]);
      acc[1][nt] = wmma_bf16(a1, b, acc[1][nt]);
    }
  }

#pragma unroll
  for (int mg = 0; mg < 2; ++mg) {
#pragma unroll
    for (int nt = 0; nt < 4; ++nt) {
      const int col = col0 + nt * 16 + (lane & 15);
      float bv = 0.f;
      if (HASBIAS) bv = (col < nvalid) ? bias[col] : 0.f;
#pragma unroll
      for (int jj = 0; jj < 8; ++jj) {
        const long row = row0 + wave * 32 + mg * 16 + hi * 8 + jj;
        float v = acc[mg][nt][jj] + bv;
        if (RELU) v = fmaxf(v, 0.f);
        if (RESID) { if (col < nvalid) v += resid[row * nvalid + col]; }
        if (OUTF32) { if (col < nvalid) outF[row * nvalid + col] = v; }
        else        { outB[row * N + col] = (__bf16)v; }
      }
    }
  }
}

// ---------------- Causal attention: one block per (b, h, 64-query slab) ----------------
// qkv: bf16 [32768][2304]; attn out: bf16 [32768][704] (concat heads, stride 114 per head)
__global__ __launch_bounds__(128) void attention_kernel(
    const __bf16* __restrict__ qkv, __bf16* __restrict__ attn)
{
  __shared__ __attribute__((aligned(16))) int    Qs[64 * 64];     // 64 q-rows x 128 e (bf16)
  __shared__ __attribute__((aligned(16))) int    Ks[16 * 64];     // 16 keys  x 128 e (bf16)
  __shared__ __attribute__((aligned(16))) __bf16 Ss[64 * 256];    // scores, then P (bf16)
  __shared__ __attribute__((aligned(16))) __bf16 Vs[128 * 32];    // [e][key] K-contig
  const int* Si = (const int*)Ss;
  const int* Vi = (const int*)Vs;

  const int tid  = threadIdx.x;
  const int lane = tid & 31, wave = tid >> 5, hi = lane >> 4;
  const int bh = blockIdx.x;
  const int b  = bh / 6, h = bh % 6;
  const int q0 = blockIdx.y * 64;
  const int kend = q0 + 64;                     // causal key extent for this slab
  const int* qg = (const int*)qkv;
  const long rowdw = 1152;                      // 2304 bf16 per token row
  const long base  = (long)(b * 256) * rowdw + h * 64;

  // load Q slab
#pragma unroll 4
  for (int i = 0; i < 32; ++i) {
    int idx = tid + i * 128;
    int r = idx >> 6, c = idx & 63;
    Qs[idx] = qg[base + (long)(q0 + r) * rowdw + c];
  }

  // ---- S = Q K^T * scale ----
  for (int j0 = 0; j0 < kend; j0 += 16) {
    __syncthreads();
#pragma unroll
    for (int i = 0; i < 8; ++i) {
      int idx = tid + i * 128;
      int r = idx >> 6, c = idx & 63;
      Ks[idx] = qg[base + 384 + (long)(j0 + r) * rowdw + c];
    }
    __syncthreads();
    v8f s = {};
#pragma unroll
    for (int ks = 0; ks < 4; ++ks) {
      v8i a, bf;
      const int abase = (wave * 16 + (lane & 15)) * 64 + ks * 16;
      const int bbase = (lane & 15) * 64 + ks * 16;
#pragma unroll
      for (int v = 0; v < 8; ++v) { a[v] = Qs[abase + fdA(v, hi)]; bf[v] = Ks[bbase + fdB(v, hi)]; }
      s = wmma_bf16(a, bf, s);
    }
    const int keyc = j0 + (lane & 15);
#pragma unroll
    for (int jj = 0; jj < 8; ++jj) {
      int r = wave * 16 + hi * 8 + jj;
      Ss[r * 256 + keyc] = (__bf16)(s[jj] * 0.09365858f);   // 114^-0.5
    }
  }
  __syncthreads();

  // ---- causal softmax per row (threads 0..63) ----
  if (tid < 64) {
    const int r = tid;
    const int limit = q0 + r;                       // inclusive
    __bf16* srow = &Ss[r * 256];
    float m = -1e30f;
    for (int c = 0; c <= limit; ++c) m = fmaxf(m, (float)srow[c]);
    float l = 0.f;
    for (int c = 0; c <= limit; ++c) {
      float e = __expf((float)srow[c] - m);
      l += e;
      srow[c] = (__bf16)e;
    }
    const float inv = 1.f / l;
    for (int c = 0; c <= limit; ++c) srow[c] = (__bf16)((float)srow[c] * inv);
    for (int c = limit + 1; c < kend; ++c) srow[c] = (__bf16)0.f;
  }

  // ---- O = P V ----
  v8f o[8] = {};
  for (int j0 = 0; j0 < kend; j0 += 32) {
    __syncthreads();
#pragma unroll
    for (int i = 0; i < 16; ++i) {                  // stage V tile [32 keys][128 e] -> [e][key]
      int idx = tid + i * 128;
      int kr = idx >> 6, cd = idx & 63;
      union { int w; __bf16 hh[2]; } u;
      u.w = qg[base + 768 + (long)(j0 + kr) * rowdw + cd];
      Vs[(2 * cd) * 32 + kr]     = u.hh[0];
      Vs[(2 * cd + 1) * 32 + kr] = u.hh[1];
    }
    __syncthreads();
    v8i a;
    const int abase = (wave * 16 + (lane & 15)) * 128 + (j0 >> 1);
#pragma unroll
    for (int v = 0; v < 8; ++v) a[v] = Si[abase + fdA(v, hi)];
#pragma unroll
    for (int nt = 0; nt < 8; ++nt) {
      v8i bf;
      const int bbase = (nt * 16 + (lane & 15)) * 16;
#pragma unroll
      for (int v = 0; v < 8; ++v) bf[v] = Vi[bbase + fdB(v, hi)];
      o[nt] = wmma_bf16(a, bf, o[nt]);
    }
  }

  // store concat-head output (only e < 114)
#pragma unroll
  for (int nt = 0; nt < 8; ++nt) {
    const int e = nt * 16 + (lane & 15);
    if (e < 114) {
#pragma unroll
      for (int jj = 0; jj < 8; ++jj) {
        const int r = wave * 16 + hi * 8 + jj;
        const long grow = (long)(b * 256 + q0 + r);
        attn[grow * 704 + h * 114 + e] = (__bf16)o[nt][jj];
      }
    }
  }
}

// ======================= host launcher =======================
extern "C" void kernel_launch(void* const* d_in, const int* in_sizes, int n_in,
                              void* d_out, int out_size, void* d_ws, size_t ws_size,
                              hipStream_t stream) {
  const float* x    = (const float*)d_in[0];
  const float* ln1g = (const float*)d_in[1];
  const float* ln1b = (const float*)d_in[2];
  const float* Wq   = (const float*)d_in[3];
  const float* Wk   = (const float*)d_in[4];
  const float* Wv   = (const float*)d_in[5];
  const float* Wo   = (const float*)d_in[6];
  const float* bo   = (const float*)d_in[7];
  const float* ln2g = (const float*)d_in[8];
  const float* ln2b = (const float*)d_in[9];
  const float* W1   = (const float*)d_in[10];
  const float* b1   = (const float*)d_in[11];
  const float* W2   = (const float*)d_in[12];
  const float* b2   = (const float*)d_in[13];

  char* ws = (char*)d_ws;
  size_t off = 0;
  auto take = [&](size_t bytes) { char* p = ws + off; off += (bytes + 255) & ~(size_t)255; return p; };
  const long Mrows = 32768;            // B*T
  __bf16* xn   = (__bf16*)take(Mrows * 704 * 2);       // LN1 out, later LN2 out
  __bf16* wqkv = (__bf16*)take(2304L * 704 * 2);       // [N][K]
  __bf16* qkv  = (__bf16*)take(Mrows * 2304 * 2);      // reused (with attn) as FFN hidden
  __bf16* attnb= (__bf16*)take(Mrows * 704 * 2);
  __bf16* hbuf = qkv;                                  // 32768x2752 fits in qkv+attn region
  __bf16* wo   = (__bf16*)take(704L * 704 * 2);        // [N][K]
  float*  x2   = (float*)take(Mrows * 684 * 4);
  __bf16* w1   = (__bf16*)take(2752L * 704 * 2);       // [N][K]
  __bf16* w2   = (__bf16*)take(704L * 2752 * 2);       // [N][K]

  // weight prep (bf16, zero-padded, transposed to [N][K])
  packqkvT_kernel<<<6336, 256, 0, stream>>>(Wq, Wk, Wv, wqkv);
  padconvT_kernel<<<1936, 256, 0, stream>>>(Wo, wo, 684, 684, 704, 704);
  padconvT_kernel<<<7568, 256, 0, stream>>>(W1, w1, 684, 2736, 704, 2752);
  padconvT_kernel<<<7568, 256, 0, stream>>>(W2, w2, 2736, 684, 2752, 704);

  // LN1
  layernorm_kernel<<<32768, 256, 0, stream>>>(x, ln1g, ln1b, xn, 684, 704);
  // QKV projection: [32768x704] x [704x2304]
  gemm_bf16_kernel<false,false,false,false><<<dim3(128, 36), 256, 0, stream>>>(
      xn, wqkv, nullptr, nullptr, nullptr, qkv, 32768, 2304, 704, 2304);
  // attention (zero attn buffer first so pad cols 684..703 are 0)
  (void)hipMemsetAsync(attnb, 0, (size_t)Mrows * 704 * 2, stream);
  attention_kernel<<<dim3(768, 4), 128, 0, stream>>>(qkv, attnb);
  // output projection + bias + residual -> x2 (fp32)
  gemm_bf16_kernel<true,false,true,true><<<dim3(128, 11), 256, 0, stream>>>(
      attnb, wo, bo, x, x2, nullptr, 32768, 704, 704, 684);
  // LN2
  layernorm_kernel<<<32768, 256, 0, stream>>>(x2, ln2g, ln2b, xn, 684, 704);
  // FFN1 + bias + relu -> h (bf16)
  gemm_bf16_kernel<true,true,false,false><<<dim3(128, 43), 256, 0, stream>>>(
      xn, w1, b1, nullptr, nullptr, hbuf, 32768, 2752, 704, 2736);
  // FFN2 + bias + residual -> d_out (fp32)
  gemm_bf16_kernel<true,false,true,true><<<dim3(128, 11), 256, 0, stream>>>(
      hbuf, w2, b2, x2, (float*)d_out, nullptr, 32768, 704, 2752, 684);

  (void)in_sizes; (void)n_in; (void)out_size; (void)ws_size;
}
